// MobaAttention_83399674953994
// MI455X (gfx1250) — matrix operand order repaired and checked by reading, not compiled, not run on hardware
//
#include <hip/hip_runtime.h>
#include <math.h>

// Problem constants (match reference)
#define S_   2048
#define D_   2048
#define H_   32
#define HD_  64
#define CS_  256
#define NCH_ 8
#define TOPK_ 4

typedef __bf16 bf16_t;
typedef __attribute__((ext_vector_type(16))) __bf16 v16bf;
typedef __attribute__((ext_vector_type(8)))  __bf16 v8bf;
typedef __attribute__((ext_vector_type(8)))  float  v8f;

#define NEGINF (-__builtin_inff())

__device__ __forceinline__ bf16_t f2bf(float f) {
  unsigned u = __builtin_bit_cast(unsigned, f);
  u += 0x7FFFu + ((u >> 16) & 1u);          // round-to-nearest-even
  unsigned short h = (unsigned short)(u >> 16);
  return __builtin_bit_cast(bf16_t, h);
}

__device__ __forceinline__ v8f wmma_bf16(v16bf a, v16bf b, v8f c) {
  // D(f32 16x16) = A(bf16 16x32) * B(bf16 32x16) + C
  return __builtin_amdgcn_wmma_f32_16x16x32_bf16(false, a, false, b, (short)0, c,
                                                 false, false);
}

// Async copy 16B/lane from global (saddr + 32-bit voffset) into LDS (vdst = LDS
// byte address). INST_OFFSET is added to BOTH global and LDS addresses.
__device__ __forceinline__ void async_g2l_b128_0(unsigned lds, unsigned goff,
                                                 const void* base) {
  asm volatile("global_load_async_to_lds_b128 %0, %1, %2 offset:0"
               :: "v"(lds), "v"(goff), "s"(base) : "memory");
}
__device__ __forceinline__ void async_g2l_b128_16(unsigned lds, unsigned goff,
                                                  const void* base) {
  asm volatile("global_load_async_to_lds_b128 %0, %1, %2 offset:16"
               :: "v"(lds), "v"(goff), "s"(base) : "memory");
}
__device__ __forceinline__ void wait_async0() {
  asm volatile("s_wait_asynccnt 0x0" ::: "memory");
}

// ---------------------------------------------------------------------------
// fp32 -> bf16 cast
// ---------------------------------------------------------------------------
__global__ void cast_f32_to_bf16(const float* __restrict__ in,
                                 bf16_t* __restrict__ out, int n) {
  int i = blockIdx.x * blockDim.x + threadIdx.x;
  if (i < n) out[i] = f2bf(in[i]);
}

// ---------------------------------------------------------------------------
// Tiled bf16 WMMA GEMM: C[M,N](f32) = A[M,K](bf16,row-major) * B[K,N](bf16,row-major)
// N, K compile-time. BM=128, BN=64, BK=32; 256 threads = 8 waves; wave tile
// 32x32 = 2x2 WMMA tiles. A staged via GLOBAL_LOAD_ASYNC_TO_LDS_B128 into a
// ping-pong LDS buffer (tile k+1 prefetched while tile k feeds WMMA).
// ---------------------------------------------------------------------------
#define BM 128
#define BN 64
#define BK 32

template <int N, int K>
__global__ __launch_bounds__(256)
void gemm_bf16_kernel(const bf16_t* __restrict__ A, const bf16_t* __restrict__ B,
                      float* __restrict__ C) {
  __shared__ __attribute__((aligned(32))) bf16_t As[2][BM][BK];   // m-major
  __shared__ __attribute__((aligned(32))) bf16_t Bs[2][BN][BK];   // transposed [n][k]

  const int tid  = threadIdx.x;
  const int lane = tid & 31;
  const int ln   = lane & 15;
  const int hi   = lane >> 4;
  const int wid  = tid >> 5;
  const int wm   = wid & 3;     // 0..3 -> rows
  const int wn   = wid >> 2;    // 0..1 -> cols
  const int bm   = blockIdx.x * BM;
  const int bn   = blockIdx.y * BN;

  // per-thread staging coordinates
  const int arow = tid >> 1;              // A: 2 threads per row, 32B each
  const int akof = (tid & 1) * 16;
  const int bkr  = tid >> 3;              // B: 0..31 (k row)
  const int bcof = (tid & 7) * 8;         // 0..56 (col group)

  const unsigned asBase = (unsigned)(size_t)(&As[0][0][0]);
  const unsigned aLds   = asBase + (unsigned)(arow * (BK * 2) + akof * 2);

  v8f acc[2][2];
#pragma unroll
  for (int i = 0; i < 2; ++i)
#pragma unroll
    for (int j = 0; j < 2; ++j)
#pragma unroll
      for (int v = 0; v < 8; ++v) acc[i][j][v] = 0.0f;

  auto stageA = [&](int k0, int buf) {
    const unsigned goff = (unsigned)(((bm + arow) * K + k0 + akof) * 2);
    const unsigned lds  = aLds + (unsigned)buf * (BM * BK * 2);
    async_g2l_b128_0(lds, goff, (const void*)A);
    async_g2l_b128_16(lds, goff, (const void*)A);
  };
  auto loadB = [&](int k0) -> v8bf {
    return *reinterpret_cast<const v8bf*>(B + (size_t)(k0 + bkr) * N + bn + bcof);
  };
  auto storeB = [&](v8bf bv, int buf) {
#pragma unroll
    for (int e = 0; e < 8; ++e) Bs[buf][bcof + e][bkr] = bv[e];
  };

  // prologue: stage tile 0
  stageA(0, 0);
  storeB(loadB(0), 0);
  wait_async0();
  __syncthreads();

  constexpr int nk = K / BK;
  int p = 0;
  for (int kt = 0; kt < nk; ++kt) {
    const bool has_next = (kt + 1 < nk);
    v8bf bnext;
    if (has_next) {
      stageA((kt + 1) * BK, p ^ 1);       // async prefetch A into other buffer
      bnext = loadB((kt + 1) * BK);       // B prefetch into registers
    }

    // fragments from buffer p
    v16bf afr[2];
#pragma unroll
    for (int i = 0; i < 2; ++i) {
      const int row = wm * 32 + i * 16 + ln;
      const int kh  = hi * 16;
      v8bf lo = *reinterpret_cast<const v8bf*>(&As[p][row][kh]);
      v8bf hb = *reinterpret_cast<const v8bf*>(&As[p][row][8 + kh]);
#pragma unroll
      for (int e = 0; e < 8; ++e) { afr[i][e] = lo[e]; afr[i][8 + e] = hb[e]; }
    }
    v16bf bfr[2];
#pragma unroll
    for (int j = 0; j < 2; ++j) {
      const int col = wn * 32 + j * 16 + ln;
      bfr[j] = *reinterpret_cast<const v16bf*>(&Bs[p][col][hi * 16]);
    }
#pragma unroll
    for (int i = 0; i < 2; ++i)
#pragma unroll
      for (int j = 0; j < 2; ++j) acc[i][j] = wmma_bf16(afr[i], bfr[j], acc[i][j]);

    if (has_next) storeB(bnext, p ^ 1);
    wait_async0();
    __syncthreads();
    p ^= 1;
  }

  // epilogue: constant-stride stores (N is constexpr -> immediate offsets)
#pragma unroll
  for (int i = 0; i < 2; ++i)
#pragma unroll
    for (int j = 0; j < 2; ++j) {
      const int row0 = bm + wm * 32 + i * 16 + hi * 8;
      float* cp = C + (size_t)row0 * N + bn + wn * 32 + j * 16 + ln;
#pragma unroll
      for (int v = 0; v < 8; ++v) cp[v * N] = acc[i][j][v];
    }
}

// ---------------------------------------------------------------------------
// RoPE: in-place on fp32 [S, H*HD], also writes bf16 copy
// ---------------------------------------------------------------------------
__global__ void rope_kernel(float* __restrict__ X, bf16_t* __restrict__ Xb) {
  int idx = blockIdx.x * blockDim.x + threadIdx.x;   // S*H*32
  if (idx >= S_ * H_ * 32) return;
  const int d = idx & 31;
  const int h = (idx >> 5) & 31;
  const int s = idx >> 10;
  const float inv = expf(-0.28782313662425572f * (float)d); // 1/10000^(d/32)
  const float fr  = (float)s * inv;
  const float c = cosf(fr), sn = sinf(fr);
  const size_t base = (size_t)s * D_ + h * HD_;
  const float x1 = X[base + d], x2 = X[base + d + 32];
  const float r1 = x1 * c - x2 * sn;
  const float r2 = x2 * c + x1 * sn;
  X[base + d]       = r1;
  X[base + d + 32]  = r2;
  Xb[base + d]      = f2bf(r1);
  Xb[base + d + 32] = f2bf(r2);
}

// ---------------------------------------------------------------------------
// Chunk means of K: kg[c][h][d] = mean over CS rows
// ---------------------------------------------------------------------------
__global__ void kgmean_kernel(const float* __restrict__ Kf, float* __restrict__ kg) {
  int idx = blockIdx.x * blockDim.x + threadIdx.x;   // NCH*H*HD
  if (idx >= NCH_ * H_ * HD_) return;
  const int d = idx & 63;
  const int h = (idx >> 6) & 31;
  const int c = idx >> 11;
  float s = 0.0f;
  for (int i = 0; i < CS_; ++i)
    s += Kf[(size_t)(c * CS_ + i) * D_ + h * HD_ + d];
  kg[idx] = s * (1.0f / CS_);
}

// ---------------------------------------------------------------------------
// Gate + top-4 chunk selection -> bitmask per (h, s)
// ---------------------------------------------------------------------------
__global__ void gate_topk_kernel(const float* __restrict__ Qf,
                                 const float* __restrict__ kg,
                                 unsigned char* __restrict__ mask) {
  int idx = blockIdx.x * blockDim.x + threadIdx.x;   // H*S, idx = h*S + s
  if (idx >= H_ * S_) return;
  const int s  = idx & (S_ - 1);
  const int h  = idx >> 11;
  const int cs = s / CS_;
  const float* q = Qf + (size_t)s * D_ + h * HD_;
  float g[NCH_];
#pragma unroll
  for (int c = 0; c < NCH_; ++c) {
    if (c > cs)       g[c] = -3.0e38f;      // future chunk -> -inf
    else if (c == cs) g[c] = 3.0e38f;       // self chunk   -> +inf
    else {
      const float* kgp = kg + (size_t)(c * H_ + h) * HD_;
      float acc = 0.0f;
      for (int d = 0; d < HD_; ++d) acc += q[d] * kgp[d];
      g[c] = acc;
    }
  }
  // stable top-4 (ties -> lowest index), matching jax.lax.top_k
  unsigned m = 0;
#pragma unroll
  for (int t = 0; t < TOPK_; ++t) {
    int best = -1;
    float bv = 0.0f;
    for (int c = 0; c < NCH_; ++c) {
      if (((m >> c) & 1u) == 0u && (best < 0 || g[c] > bv)) { best = c; bv = g[c]; }
    }
    m |= 1u << best;
  }
  mask[idx] = (unsigned char)m;
}

// ---------------------------------------------------------------------------
// MoBA flash attention: one wave per (head, 16-row q tile)
// ---------------------------------------------------------------------------
__global__ __launch_bounds__(32)
void moba_attn_kernel(const bf16_t* __restrict__ Qb, const bf16_t* __restrict__ Kb,
                      const bf16_t* __restrict__ Vb,
                      const unsigned char* __restrict__ mask,
                      const float* __restrict__ G, const float* __restrict__ onw,
                      bf16_t* __restrict__ Ob) {
  __shared__ __attribute__((aligned(32))) bf16_t Pb[16][32];
  const int lane = threadIdx.x;
  const int ln   = lane & 15;
  const int hi   = lane >> 4;
  const int h    = blockIdx.x;
  const int s0   = blockIdx.y * 16;
  const float scale = 0.125f;   // 1/sqrt(64)

  // Q A-fragments (K-dim = head dim, 2 steps of 32)
  v16bf aq[2];
  {
    const bf16_t* qp = Qb + (size_t)(s0 + ln) * D_ + h * HD_;
    const int kh = hi * 16;
#pragma unroll
    for (int kf = 0; kf < 2; ++kf) {
      v8bf lo = *reinterpret_cast<const v8bf*>(qp + kf * 32 + kh);
      v8bf hb = *reinterpret_cast<const v8bf*>(qp + kf * 32 + 8 + kh);
#pragma unroll
      for (int e = 0; e < 8; ++e) { aq[kf][e] = lo[e]; aq[kf][8 + e] = hb[e]; }
    }
  }

  // per-lane row chunk masks (this lane's 8 rows: hi*8 + v)
  unsigned rmask[8];
  unsigned comb = 0;
#pragma unroll
  for (int v = 0; v < 8; ++v) {
    rmask[v] = mask[h * S_ + s0 + hi * 8 + v];
    comb |= rmask[v];
  }
#pragma unroll
  for (int off = 16; off; off >>= 1)
    comb |= (unsigned)__shfl_xor((int)comb, off, 32);

  float mrow[8], lrow[8];
  v8f oacc[4];
#pragma unroll
  for (int v = 0; v < 8; ++v) { mrow[v] = NEGINF; lrow[v] = 0.0f; }
#pragma unroll
  for (int j = 0; j < 4; ++j)
#pragma unroll
    for (int v = 0; v < 8; ++v) oacc[j][v] = 0.0f;

  const int cSelf = s0 / CS_;
  for (int c = 0; c <= cSelf; ++c) {
    if (!((comb >> c) & 1u)) continue;
    const int t0 = c * CS_;
    for (int kb = 0; kb < CS_ / 32; ++kb) {
      const int tk = t0 + kb * 32;
      if (tk > s0 + 15) break;   // fully beyond causal frontier (uniform branch)

      float p[2][8];
      float sc[2][8];
      // scores for two 16-key groups
#pragma unroll
      for (int g2 = 0; g2 < 2; ++g2) {
        const int tg = tk + g2 * 16;
        v8f sf;
#pragma unroll
        for (int v = 0; v < 8; ++v) sf[v] = 0.0f;
#pragma unroll
        for (int kf = 0; kf < 2; ++kf) {
          const int key = tg + ln;    // lane's column
          v16bf bk = *reinterpret_cast<const v16bf*>(
              Kb + (size_t)key * D_ + h * HD_ + kf * 32 + hi * 16);
          sf = wmma_bf16(aq[kf], bk, sf);
        }
#pragma unroll
        for (int v = 0; v < 8; ++v) {
          const int sg = s0 + hi * 8 + v;
          const int kt = tg + ln;
          const bool ok = (((rmask[v] >> c) & 1u) != 0u) && (kt <= sg);
          sc[g2][v] = ok ? sf[v] * scale : NEGINF;
        }
      }
      // online softmax (row reductions over 16-lane halves)
#pragma unroll
      for (int v = 0; v < 8; ++v) {
        float mx = fmaxf(sc[0][v], sc[1][v]);
#pragma unroll
        for (int off = 8; off; off >>= 1) mx = fmaxf(mx, __shfl_xor(mx, off, 32));
        const float mn = fmaxf(mrow[v], mx);
        const float alpha = (mn == NEGINF) ? 1.0f : __expf(mrow[v] - mn);
        const float p0 = (sc[0][v] == NEGINF) ? 0.0f : __expf(sc[0][v] - mn);
        const float p1 = (sc[1][v] == NEGINF) ? 0.0f : __expf(sc[1][v] - mn);
        p[0][v] = p0; p[1][v] = p1;
        float rs = p0 + p1;
#pragma unroll
        for (int off = 8; off; off >>= 1) rs += __shfl_xor(rs, off, 32);
        lrow[v] = lrow[v] * alpha + rs;
        mrow[v] = mn;
#pragma unroll
        for (int j = 0; j < 4; ++j) oacc[j][v] *= alpha;
      }
      // reshape P (C-frag layout) -> A-frag layout via LDS
      __syncthreads();
#pragma unroll
      for (int v = 0; v < 8; ++v) {
        Pb[hi * 8 + v][ln]      = f2bf(p[0][v]);
        Pb[hi * 8 + v][16 + ln] = f2bf(p[1][v]);
      }
      __syncthreads();
      v16bf pa;
      {
        const int kh = hi * 8;
        v8bf lo = *reinterpret_cast<const v8bf*>(&Pb[ln][kh]);
        v8bf hb = *reinterpret_cast<const v8bf*>(&Pb[ln][16 + kh]);
#pragma unroll
        for (int e = 0; e < 8; ++e) { pa[e] = lo[e]; pa[8 + e] = hb[e]; }
      }
      // PV: B = V[key][d], 4 d-groups of 16
#pragma unroll
      for (int j = 0; j < 4; ++j) {
        v16bf vb;
        const int dcol = j * 16 + ln;
#pragma unroll
        for (int e = 0; e < 16; ++e) {
          const int key = tk + (e < 8 ? hi * 8 + e : 8 + hi * 8 + e);
          vb[e] = Vb[(size_t)key * D_ + h * HD_ + dcol];
        }
        oacc[j] = wmma_bf16(pa, vb, oacc[j]);
      }
      __syncthreads();
    }
  }

  // finalize: /l, RMS-norm over head dim, * o_norm_w, * sigmoid(g)
  float rn[8];
#pragma unroll
  for (int v = 0; v < 8; ++v) {
    const float linv = lrow[v] > 0.0f ? 1.0f / lrow[v] : 0.0f;
    float ss = 0.0f;
#pragma unroll
    for (int j = 0; j < 4; ++j) {
      const float o = oacc[j][v] * linv;
      oacc[j][v] = o;
      ss += o * o;
    }
#pragma unroll
    for (int off = 8; off; off >>= 1) ss += __shfl_xor(ss, off, 32);
    rn[v] = rsqrtf(ss * (1.0f / HD_) + 1e-6f);
  }
#pragma unroll
  for (int j = 0; j < 4; ++j) {
    const int d = j * 16 + ln;
    const float w = onw[d];
#pragma unroll
    for (int v = 0; v < 8; ++v) {
      const int sg = s0 + hi * 8 + v;
      const float gv  = G[(size_t)sg * D_ + h * HD_ + d];
      const float sig = 1.0f / (1.0f + __expf(-gv));
      Ob[(size_t)sg * D_ + h * HD_ + d] = f2bf(oacc[j][v] * rn[v] * w * sig);
    }
  }
}

// ---------------------------------------------------------------------------
// host launcher
// ---------------------------------------------------------------------------
extern "C" void kernel_launch(void* const* d_in, const int* in_sizes, int n_in,
                              void* d_out, int out_size, void* d_ws, size_t ws_size,
                              hipStream_t stream) {
  (void)in_sizes; (void)n_in; (void)out_size; (void)ws_size;
  const float* x   = (const float*)d_in[0];
  const float* Wq  = (const float*)d_in[1];
  const float* Wk  = (const float*)d_in[2];
  const float* Wv  = (const float*)d_in[3];
  const float* Wo  = (const float*)d_in[4];
  const float* Wg1 = (const float*)d_in[5];
  const float* Wg2 = (const float*)d_in[6];
  const float* onw = (const float*)d_in[7];
  float* out = (float*)d_out;

  char* p = (char*)d_ws;
  auto alloc = [&](size_t bytes) -> void* {
    void* r = (void*)p;
    p += (bytes + 255) & ~(size_t)255;
    return r;
  };
  bf16_t* xb   = (bf16_t*)alloc((size_t)S_ * D_ * 2);
  bf16_t* wqb  = (bf16_t*)alloc((size_t)D_ * D_ * 2);
  bf16_t* wkb  = (bf16_t*)alloc((size_t)D_ * D_ * 2);
  bf16_t* wvb  = (bf16_t*)alloc((size_t)D_ * D_ * 2);
  bf16_t* wob  = (bf16_t*)alloc((size_t)D_ * D_ * 2);
  bf16_t* wg1b = (bf16_t*)alloc((size_t)D_ * HD_ * 2);
  bf16_t* wg2b = (bf16_t*)alloc((size_t)HD_ * D_ * 2);
  float*  Qf   = (float*)alloc((size_t)S_ * D_ * 4);
  float*  Kf   = (float*)alloc((size_t)S_ * D_ * 4);
  float*  Vf   = (float*)alloc((size_t)S_ * D_ * 4);
  bf16_t* Qb   = (bf16_t*)alloc((size_t)S_ * D_ * 2);
  bf16_t* Kb   = (bf16_t*)alloc((size_t)S_ * D_ * 2);
  bf16_t* Vb   = (bf16_t*)alloc((size_t)S_ * D_ * 2);
  float*  Gt   = (float*)alloc((size_t)S_ * HD_ * 4);
  bf16_t* Gtb  = (bf16_t*)alloc((size_t)S_ * HD_ * 2);
  float*  Gg   = (float*)alloc((size_t)S_ * D_ * 4);
  float*  kg   = (float*)alloc((size_t)NCH_ * H_ * HD_ * 4);
  unsigned char* mk = (unsigned char*)alloc((size_t)H_ * S_);
  bf16_t* Ob   = (bf16_t*)alloc((size_t)S_ * D_ * 2);

  const int CT = 256;
  auto cast = [&](const float* i, bf16_t* o, int n) {
    cast_f32_to_bf16<<<(n + CT - 1) / CT, CT, 0, stream>>>(i, o, n);
  };
  cast(x,   xb,   S_ * D_);
  cast(Wq,  wqb,  D_ * D_);
  cast(Wk,  wkb,  D_ * D_);
  cast(Wv,  wvb,  D_ * D_);
  cast(Wo,  wob,  D_ * D_);
  cast(Wg1, wg1b, D_ * HD_);
  cast(Wg2, wg2b, HD_ * D_);

  // GEMMs (N, K compile-time)
  {
    dim3 grid(S_ / BM, D_ / BN);
    gemm_bf16_kernel<D_, D_><<<grid, dim3(256), 0, stream>>>(xb, wqb, Qf);
    gemm_bf16_kernel<D_, D_><<<grid, dim3(256), 0, stream>>>(xb, wkb, Kf);
    gemm_bf16_kernel<D_, D_><<<grid, dim3(256), 0, stream>>>(xb, wvb, Vf);
  }
  {
    dim3 grid(S_ / BM, HD_ / BN);
    gemm_bf16_kernel<HD_, D_><<<grid, dim3(256), 0, stream>>>(xb, wg1b, Gt);
  }
  cast(Gt, Gtb, S_ * HD_);
  {
    dim3 grid(S_ / BM, D_ / BN);
    gemm_bf16_kernel<D_, HD_><<<grid, dim3(256), 0, stream>>>(Gtb, wg2b, Gg);
  }

  rope_kernel<<<(S_ * H_ * 32 + CT - 1) / CT, CT, 0, stream>>>(Qf, Qb);
  rope_kernel<<<(S_ * H_ * 32 + CT - 1) / CT, CT, 0, stream>>>(Kf, Kb);
  cast(Vf, Vb, S_ * D_);
  kgmean_kernel<<<(NCH_ * H_ * HD_ + CT - 1) / CT, CT, 0, stream>>>(Kf, kg);
  gate_topk_kernel<<<(H_ * S_ + CT - 1) / CT, CT, 0, stream>>>(Qf, kg, mk);

  dim3 agrid(H_, S_ / 16);
  moba_attn_kernel<<<agrid, dim3(32), 0, stream>>>(Qb, Kb, Vb, mk, Gg, onw, Ob);

  {
    dim3 grid(S_ / BM, D_ / BN);
    gemm_bf16_kernel<D_, D_><<<grid, dim3(256), 0, stream>>>(Ob, wob, out);
  }
}
